// RNN_89850715832466
// MI455X (gfx1250) — matrix operand order repaired and checked by reading, not compiled, not run on hardware
//
#include <hip/hip_runtime.h>
#include <hip/hip_bf16.h>

// ---------------------------------------------------------------------------
// ENAS RNN on gfx1250: bf16 WMMA (f32 accumulate) for all GEMMs.
// Weights converted once per launch into padded bf16 workspace buffers
// (H 1000->1024, E+H 2000->2048). Zero padding is value-preserving through
// the sigmoid/tanh gate algebra (pad lanes of h-state can become nonzero, but only meet zero weight columns).
// Decode GEMM stages its B tile into LDS via the Tensor Data Mover (TDM).
// ---------------------------------------------------------------------------

typedef __attribute__((ext_vector_type(16))) __bf16 v16bf;
typedef __attribute__((ext_vector_type(8)))  __bf16 v8bf;
typedef __attribute__((ext_vector_type(8)))  float  v8f;
typedef __attribute__((ext_vector_type(4)))  unsigned int u32x4;
typedef __attribute__((ext_vector_type(8)))  int i32x8;
typedef __attribute__((ext_vector_type(4)))  int i32x4;

namespace {
constexpr int Bsz = 64;      // batch
constexpr int Tsz = 64;      // timesteps
constexpr int H   = 1000;    // hidden
constexpr int HP  = 1024;    // padded hidden
constexpr int EH  = 2000;    // E + H
constexpr int EHP = 2048;    // padded E + H
constexpr int V   = 10000;   // vocab
constexpr int NE  = 11;      // edges
}

// ---------------- WMMA helpers (ISA 7.12.2 layouts) -------------------------

__device__ inline v8f wmma_bf16(v16bf a, v16bf b, v8f c) {
  return __builtin_amdgcn_wmma_f32_16x16x32_bf16(
      /*neg_a=*/false, a, /*neg_b=*/false, b,
      /*c_mod=*/(short)0, c, /*reuse_a=*/false, /*reuse_b=*/false);
}

// A fragment: 16x32 bf16 tile. Lane L: row = m0 + (L&15); lanes 0-15 hold
// K = {k0..k0+7, k0+16..k0+23}, lanes 16-31 hold K = {k0+8.., k0+24..}.
__device__ inline v16bf load_frag_a(const __bf16* base, int lda,
                                    int m0, int k0, int lane) {
  const int m  = m0 + (lane & 15);
  const int kh = (lane >> 4) * 8;
  const __bf16* p = base + (size_t)m * lda + k0 + kh;
  v8bf lo = *reinterpret_cast<const v8bf*>(p);
  v8bf hi = *reinterpret_cast<const v8bf*>(p + 16);
  v16bf r;
#pragma unroll
  for (int i = 0; i < 8; ++i) { r[i] = lo[i]; r[i + 8] = hi[i]; }
  return r;
}

// B fragment: 32x16 bf16 tile of W^T, i.e. lane L supplies output column
// n = n0 + (L&15) = one row of row-major W; lanes 0-15 hold
// K = {k0..k0+7, k0+16..k0+23}, lanes 16-31 hold K = {k0+8.., k0+24..}.
__device__ inline v16bf load_frag_b(const __bf16* base, int ldb,
                                    int n0, int k0, int lane) {
  const int n = n0 + (lane & 15);
  const int k = k0 + ((lane >> 4) << 3);
  const __bf16* p = base + (size_t)n * ldb + k;
  v8bf lo = *reinterpret_cast<const v8bf*>(p);
  v8bf hi = *reinterpret_cast<const v8bf*>(p + 16);
  v16bf r;
#pragma unroll
  for (int i = 0; i < 8; ++i) { r[i] = lo[i]; r[i + 8] = hi[i]; }
  return r;
}

__device__ inline float sigm(float x) { return 1.0f / (1.0f + __expf(-x)); }

__device__ inline float apply_act(int a, float x) {
  switch (a) {
    case 0:  return fmaxf(x, 0.0f);  // relu
    case 1:  return tanhf(x);        // tanh
    case 2:  return sigm(x);         // sigmoid
    default: return x;               // identity
  }
}

// ---------------- Tensor Data Mover: stage a [16 x 1024] bf16 tile ---------
// D# per ISA ch.8: group0 = {count=1 | lds_addr | global_addr | type=2},
// group1 = {data_size=2B, tensor_dim0=1024, tensor_dim1=V, tile_dim0=1024,
//           tile_dim1=16, tensor_dim0_stride=1024}; groups 2/3 zero (2-D).
__device__ inline void tdm_load_b_tile(const __bf16* gsrc, unsigned lds_off) {
  const unsigned long long ga = (unsigned long long)(uintptr_t)gsrc;
  u32x4 g0 = { 1u,                                   // count=1, user mode
               lds_off,                              // lds_addr (bytes)
               (unsigned)ga,                         // global_addr[31:0]
               (unsigned)(ga >> 32) | (2u << 30) };  // ga[56:32] | type=2
  i32x8 g1 = { (int)(1u << 16),        // data_size = 1 (2 bytes)
               (int)(1024u << 16),     // tensor_dim0[15:0] = 1024
               (int)(10000u << 16),    // tensor_dim0 hi=0 | tensor_dim1 lo=V
               (int)(1024u << 16),     // tensor_dim1 hi=0 | tile_dim0 = 1024
               16,                     // tile_dim1 = 16 (| tile_dim2 = 0)
               1024,                   // tensor_dim0_stride[31:0] = 1024
               0, 0 };
  i32x4 z4 = {0, 0, 0, 0};
#if __clang_major__ >= 23
  i32x8 z8 = {0, 0, 0, 0, 0, 0, 0, 0};
  __builtin_amdgcn_tensor_load_to_lds(g0, g1, z4, z4, z8, 0);
#else
  __builtin_amdgcn_tensor_load_to_lds(g0, g1, z4, z4, 0);
#endif
}

// ---------------- conversion / setup kernels --------------------------------

// Generic fp32 [R,C] -> bf16 [Rp,Cp] with zero pad; batched via blockIdx.y.
__global__ void k_conv_pad(const float* __restrict__ src, __bf16* __restrict__ dst,
                           int R, int C, int Rp, int Cp) {
  const float* s = src + (size_t)blockIdx.y * R * C;
  __bf16*      d = dst + (size_t)blockIdx.y * Rp * Cp;
  const int total = Rp * Cp;
  for (int idx = blockIdx.x * blockDim.x + threadIdx.x; idx < total;
       idx += gridDim.x * blockDim.x) {
    const int r = idx / Cp, c = idx % Cp;
    d[idx] = (r < R && c < C) ? (__bf16)s[(size_t)r * C + c] : (__bf16)0.0f;
  }
}

// w_x* fp32 [1000, 2000] -> bf16 [1024, 2048]; cols 0..999 -> 0..999 (x part),
// cols 1000..1999 -> 1024..2023 (h part), rest zero.
__global__ void k_conv_wx(const float* __restrict__ src, __bf16* __restrict__ dst) {
  const int total = HP * EHP;
  for (int idx = blockIdx.x * blockDim.x + threadIdx.x; idx < total;
       idx += gridDim.x * blockDim.x) {
    const int r = idx >> 11, c = idx & (EHP - 1);
    float v = 0.0f;
    if (r < H) {
      if (c < H) v = src[(size_t)r * EH + c];
      else if (c >= HP) {
        const int sc = c - 24;           // c-1024+1000
        if (sc < EH) v = src[(size_t)r * EH + sc];
      }
    }
    dst[idx] = (__bf16)v;
  }
}

// hidden fp32 [64,1000] -> bf16 [64,1024] (initial carry h[11]).
__global__ void k_init_h(const float* __restrict__ hidden, __bf16* __restrict__ h11) {
  const int idx = blockIdx.x * blockDim.x + threadIdx.x;  // 64*1024
  const int b = idx >> 10, j = idx & (HP - 1);
  h11[idx] = (j < H) ? (__bf16)hidden[(size_t)b * H + j] : (__bf16)0.0f;
}

// Build xh = [ embed(tokens) | h_prev ] in padded bf16 [64, 2048].
__global__ void k_build_x(const int* __restrict__ tokens,
                          const float* __restrict__ emb,
                          const __bf16* __restrict__ hprev,
                          __bf16* __restrict__ xh) {
  const int idx = blockIdx.x * blockDim.x + threadIdx.x;  // 64*1024
  const int b = idx >> 10, j = idx & (HP - 1);
  const int tok = tokens[b];
  xh[(size_t)b * EHP + j] =
      (j < H) ? (__bf16)emb[(size_t)tok * H + j] : (__bf16)0.0f;
  xh[(size_t)b * EHP + HP + j] = hprev[idx];
}

// ---------------- recurrent WMMA kernels ------------------------------------

// Cell: c0 = sig(xh@Wxc^T + bxc); h0 = c0*tanh(xh@Wxh^T + bxh) + (1-c0)*hprev.
// block (32,4): 4 waves along N; grid (16, 4): 64 N-tiles x 4 M-tiles.
__global__ void __launch_bounds__(128)
k_cell(const __bf16* __restrict__ xh,
       const __bf16* __restrict__ wxc, const __bf16* __restrict__ wxh,
       const float* __restrict__ bxc, const float* __restrict__ bxh,
       float* __restrict__ c0buf, __bf16* __restrict__ h0) {
  const int lane = threadIdx.x;
  const int n0 = (blockIdx.x * blockDim.y + threadIdx.y) * 16;
  const int m0 = blockIdx.y * 16;
  v8f accC = {}; v8f accH = {};
  for (int k0 = 0; k0 < EHP; k0 += 32) {
    v16bf a  = load_frag_a(xh, EHP, m0, k0, lane);
    v16bf bc = load_frag_b(wxc, EHP, n0, k0, lane);
    v16bf bh = load_frag_b(wxh, EHP, n0, k0, lane);
    accC = wmma_bf16(a, bc, accC);
    accH = wmma_bf16(a, bh, accH);
  }
  const int n = n0 + (lane & 15);
  const float bc = (n < H) ? bxc[n] : 0.0f;
  const float bh = (n < H) ? bxh[n] : 0.0f;
#pragma unroll
  for (int r = 0; r < 8; ++r) {
    const int m = m0 + ((lane >> 4) << 3) + r;
    const float hp = (float)xh[(size_t)m * EHP + HP + n];
    const float c  = sigm(accC[r] + bc);
    const float h  = c * tanhf(accH[r] + bh) + (1.0f - c) * hp;
    c0buf[(size_t)m * HP + n] = c;
    h0[(size_t)m * HP + n]    = (__bf16)h;
  }
}

// Edge e=(i,j): cj = sig(h_i@Wc^T); h_j = cj*act(h_i@Wh^T) + (1-c0)*h_i.
__global__ void __launch_bounds__(128)
k_edge(const __bf16* __restrict__ hsrc, __bf16* __restrict__ hdst,
       const __bf16* __restrict__ Wc, const __bf16* __restrict__ Wh,
       const float* __restrict__ c0buf, int act) {
  const int lane = threadIdx.x;
  const int n0 = (blockIdx.x * blockDim.y + threadIdx.y) * 16;
  const int m0 = blockIdx.y * 16;
  v8f accC = {}; v8f accH = {};
  for (int k0 = 0; k0 < HP; k0 += 32) {
    v16bf a  = load_frag_a(hsrc, HP, m0, k0, lane);
    v16bf bc = load_frag_b(Wc, HP, n0, k0, lane);
    v16bf bh = load_frag_b(Wh, HP, n0, k0, lane);
    accC = wmma_bf16(a, bc, accC);
    accH = wmma_bf16(a, bh, accH);
  }
  const int n = n0 + (lane & 15);
#pragma unroll
  for (int r = 0; r < 8; ++r) {
    const int m = m0 + ((lane >> 4) << 3) + r;
    const float hi = (float)hsrc[(size_t)m * HP + n];
    const float cj = sigm(accC[r]);
    const float c0 = c0buf[(size_t)m * HP + n];
    const float hj = cj * apply_act(act, accH[r]) + (1.0f - c0) * hi;
    hdst[(size_t)m * HP + n] = (__bf16)hj;
  }
}

// Mean of the 6 leaf states -> bf16 output slab for this timestep.
__global__ void k_leaf_mean(const __bf16* __restrict__ h5, const __bf16* __restrict__ h7,
                            const __bf16* __restrict__ h8, const __bf16* __restrict__ h9,
                            const __bf16* __restrict__ h10, const __bf16* __restrict__ h11,
                            __bf16* __restrict__ out) {
  const int idx = blockIdx.x * blockDim.x + threadIdx.x;  // 64*1024
  const float s = (float)h5[idx] + (float)h7[idx] + (float)h8[idx] +
                  (float)h9[idx] + (float)h10[idx] + (float)h11[idx];
  out[idx] = (__bf16)(s * (1.0f / 6.0f));
}

// Decode: [4096, 1024]bf16 @ dec_w^T [1024, 10000] + dec_b -> fp32 d_out.
// block (32,4): 4 waves along M share one TDM-staged B tile in LDS.
// grid (625, 64).
__global__ void __launch_bounds__(128)
k_decode(const __bf16* __restrict__ A, const __bf16* __restrict__ Bw,
         const float* __restrict__ bias, float* __restrict__ out) {
  __shared__ __bf16 smem[16 * HP];  // 32 KB: B rows n0..n0+15, full K
  const int lane = threadIdx.x;
  const int n0 = blockIdx.x * 16;

  // Wave 0 DMAs the B tile into LDS (TENSORcnt is per-wave, so the issuing
  // wave waits, then the workgroup barrier publishes the tile).
  if (threadIdx.y == 0) {
    tdm_load_b_tile(Bw + (size_t)n0 * HP, (unsigned)(uintptr_t)smem);
    __builtin_amdgcn_s_wait_tensorcnt(0);
  }
  __syncthreads();

  const int m0 = (blockIdx.y * blockDim.y + threadIdx.y) * 16;
  v8f acc = {};
  for (int k0 = 0; k0 < HP; k0 += 32) {
    __builtin_prefetch(A + (size_t)(m0 + (lane & 15)) * HP + k0 + 256, 0, 1);
    v16bf a = load_frag_a(A, HP, m0, k0, lane);
    v16bf b = load_frag_b(smem, HP, 0, k0, lane);   // from LDS (ds loads)
    acc = wmma_bf16(a, b, acc);
  }
  const int n = n0 + (lane & 15);
  const float bb = bias[n];
#pragma unroll
  for (int r = 0; r < 8; ++r) {
    const int m = m0 + ((lane >> 4) << 3) + r;
    out[(size_t)m * V + n] = acc[r] + bb;
  }
}

// ---------------------------------------------------------------------------

extern "C" void kernel_launch(void* const* d_in, const int* in_sizes, int n_in,
                              void* d_out, int out_size, void* d_ws, size_t ws_size,
                              hipStream_t stream) {
  (void)in_sizes; (void)n_in; (void)out_size; (void)ws_size;

  const int*   inputs = (const int*)  d_in[0];   // [T, B]
  const float* hidden = (const float*)d_in[1];   // [B, H]
  const float* emb    = (const float*)d_in[2];   // [V, H]
  const float* w_xh   = (const float*)d_in[3];   // [H, E+H]
  const float* b_xh   = (const float*)d_in[4];   // [H]
  const float* w_xc   = (const float*)d_in[5];   // [H, E+H]
  const float* b_xc   = (const float*)d_in[6];   // [H]
  const float* Wh     = (const float*)d_in[7];   // [11, H, H]
  const float* Wc     = (const float*)d_in[8];   // [11, H, H]
  const float* dec_w  = (const float*)d_in[9];   // [V, H]
  const float* dec_b  = (const float*)d_in[10];  // [V]
  float* out = (float*)d_out;                    // [T, B, V]

  // ---- carve workspace (≈82 MB) ----
  char* p = (char*)d_ws;
  auto carve = [&](size_t bytes) -> char* {
    char* r = p; p += (bytes + 255) & ~(size_t)255; return r;
  };
  __bf16* wxh_p = (__bf16*)carve((size_t)HP * EHP * 2);          // 4 MB
  __bf16* wxc_p = (__bf16*)carve((size_t)HP * EHP * 2);          // 4 MB
  __bf16* Wh_p  = (__bf16*)carve((size_t)NE * HP * HP * 2);      // 22 MB
  __bf16* Wc_p  = (__bf16*)carve((size_t)NE * HP * HP * 2);      // 22 MB
  __bf16* dec_p = (__bf16*)carve((size_t)V * HP * 2);            // 20 MB
  __bf16* xh    = (__bf16*)carve((size_t)Bsz * EHP * 2);
  float*  c0b   = (float*) carve((size_t)Bsz * HP * 4);
  __bf16* hbuf  = (__bf16*)carve((size_t)12 * Bsz * HP * 2);
  __bf16* outbf = (__bf16*)carve((size_t)Tsz * Bsz * HP * 2);    // 8 MB

  auto hstate = [&](int i) { return hbuf + (size_t)i * Bsz * HP; };

  // ---- one-time weight conversion to padded bf16 ----
  k_conv_wx<<<dim3(2048), dim3(256), 0, stream>>>(w_xh, wxh_p);
  k_conv_wx<<<dim3(2048), dim3(256), 0, stream>>>(w_xc, wxc_p);
  k_conv_pad<<<dim3(2048, NE), dim3(256), 0, stream>>>(Wh, Wh_p, H, H, HP, HP);
  k_conv_pad<<<dim3(2048, NE), dim3(256), 0, stream>>>(Wc, Wc_p, H, H, HP, HP);
  k_conv_pad<<<dim3(8192, 1), dim3(256), 0, stream>>>(dec_w, dec_p, V, H, V, HP);
  k_init_h<<<dim3(Bsz * HP / 256), dim3(256), 0, stream>>>(hidden, hstate(11));

  // ---- DAG description (BFS order => stream order satisfies deps) ----
  static const int esrc[NE] = {0, 0, 1, 1, 2, 2, 3, 3, 4, 4, 6};
  static const int edst[NE] = {1, 6, 2, 8, 3, 9, 4, 10, 5, 11, 7};
  static const int eact[NE] = {0, 1, 0, 2, 1, 0, 3, 1, 0, 2, 1};  // relu,tanh,...

  const dim3 gTile(16, 4), bTile(32, 4);  // 64 N-tiles x 4 M-tiles, 4 waves/block

  for (int t = 0; t < Tsz; ++t) {
    k_build_x<<<dim3(Bsz * HP / 256), dim3(256), 0, stream>>>(
        inputs + (size_t)t * Bsz, emb, hstate(11), xh);

    k_cell<<<gTile, bTile, 0, stream>>>(xh, wxc_p, wxh_p, b_xc, b_xh,
                                        c0b, hstate(0));

    for (int e = 0; e < NE; ++e) {
      k_edge<<<gTile, bTile, 0, stream>>>(
          hstate(esrc[e]), hstate(edst[e]),
          Wc_p + (size_t)e * HP * HP, Wh_p + (size_t)e * HP * HP,
          c0b, eact[e]);
    }

    k_leaf_mean<<<dim3(Bsz * HP / 256), dim3(256), 0, stream>>>(
        hstate(5), hstate(7), hstate(8), hstate(9), hstate(10), hstate(11),
        outbf + (size_t)t * Bsz * HP);
  }

  // ---- decode: [4096,1024] @ dec^T -> [4096, 10000] fp32 ----
  k_decode<<<dim3(V / 16, 64), dim3(32, 4), 0, stream>>>(outbf, dec_p, dec_b, out);
}
